// Multi_transformer_59407987638953
// MI455X (gfx1250) — matrix operand order repaired and checked by reading, not compiled, not run on hardware
//
#include <hip/hip_runtime.h>

// ---------------------------------------------------------------------------
// MI455X (gfx1250, wave32) implementation of the Multi-transformer block.
// All 1x1 convs are WMMA f32_16x16x32_f16 GEMMs (fp16 in, fp32 accum).
// Intermediates fp16 (bandwidth-bound problem @ 23.3 TB/s HBM).
// ---------------------------------------------------------------------------

typedef _Float16 h16;
typedef __attribute__((ext_vector_type(16))) _Float16 v16h;
typedef __attribute__((ext_vector_type(8)))  _Float16 v8h;
typedef __attribute__((ext_vector_type(8)))  float    v8f;
typedef __attribute__((ext_vector_type(4)))  float    v4f;

constexpr int BATCH = 4;
constexpr int CCH   = 64;        // channels
constexpr int HH    = 256;
constexpr int WW    = 256;
constexpr int NPIX  = HH * WW;          // 65536
constexpr int MTOT  = BATCH * NPIX;     // 262144 pixels total
constexpr int NHEAD = 8;
constexpr int CHD   = 8;                // per-head dim
constexpr int HIDD  = 170;

// ---------------------------------------------------------------------------
// Generic fp32 -> fp16 weight convert with zero padding (rows x cols).
// ---------------------------------------------------------------------------
__global__ __launch_bounds__(256) void k_cvt_pad(const float* __restrict__ src,
                                                 h16* __restrict__ dst,
                                                 int srows, int scols,
                                                 int drows, int dcols) {
  int i = blockIdx.x * 256 + threadIdx.x;
  if (i >= drows * dcols) return;
  int r = i / dcols, c = i % dcols;
  dst[i] = (r < srows && c < scols) ? (h16)src[r * scols + c] : (h16)0.0f;
}

__global__ __launch_bounds__(256) void k_zero_f32(float* __restrict__ p, int n) {
  int i = blockIdx.x * 256 + threadIdx.x;
  if (i < n) p[i] = 0.0f;
}

// ---------------------------------------------------------------------------
// Bias-free LayerNorm over channel dim (var only), pixel-major in/out.
// in:  (MTOT, 64) f32   out: (MTOT, 64) f16
// ---------------------------------------------------------------------------
__global__ __launch_bounds__(256) void k_ln(const float* __restrict__ x,
                                            const float* __restrict__ w,
                                            h16* __restrict__ out) {
  __shared__ float sw[CCH];
  if (threadIdx.x < CCH) sw[threadIdx.x] = w[threadIdx.x];
  __syncthreads();
  size_t m = (size_t)blockIdx.x * 256 + threadIdx.x;
  const float* xr = x + m * CCH;
  float vbuf[CCH];
  float s = 0.f, s2 = 0.f;
#pragma unroll
  for (int c = 0; c < CCH; c += 4) {
    v4f v = *(const v4f*)(xr + c);
#pragma unroll
    for (int i = 0; i < 4; i++) {
      float t = v[i];
      vbuf[c + i] = t;
      s += t;
      s2 += t * t;
    }
  }
  float mean = s * (1.0f / CCH);
  float var  = s2 * (1.0f / CCH) - mean * mean;
  float inv  = rsqrtf(var + 1e-5f);
  h16* o = out + m * CCH;
#pragma unroll
  for (int c = 0; c < CCH; c++) o[c] = (h16)(vbuf[c] * inv * sw[c]);
}

// ---------------------------------------------------------------------------
// WMMA GEMM:  D[m,n] = sum_k A[m,k] * W[n,k]   (+ epilogue variant)
// A: (MTOT, KPAD) f16 pixel-major.  W: (coutP, KPAD) f16 (LDS-staged).
// EPI 0: planar f16 out  (out plane n, element m)           -> conv consumers
// EPI 1: out f32 pixel-major xi[m*64+n] = acc + res[m*64+n] -> proj + residual
// EPI 2: final NCHW f32 out + residual xi                   -> pout epilogue
// Block = 256 threads = 8 waves; each wave owns a 16-row strip; tile N=16.
// ---------------------------------------------------------------------------
template <int KPAD, int EPI>
__global__ __launch_bounds__(256) void k_gemm_wmma(
    const h16* __restrict__ A, const h16* __restrict__ Wt, int coutP, int cout,
    h16* __restrict__ outP, float* __restrict__ outF,
    const float* __restrict__ res) {
  constexpr int KF = KPAD / 32;
  __shared__ __align__(32) h16 sW[22528];  // up to 352x64 halves (45 KB)

  // Stage weights to LDS (coalesced, vectorized).
  int tid = threadIdx.x;
  int total = coutP * KPAD;
  for (int i = tid * 8; i < total; i += 256 * 8)
    *(v8h*)(sW + i) = *(const v8h*)(Wt + i);
  __syncthreads();

  int wave  = tid >> 5;
  int lane  = tid & 31;
  int lhalf = lane >> 4;   // 0: lanes 0-15, 1: lanes 16-31
  int lmod  = lane & 15;
  int mbase = blockIdx.x * 128 + wave * 16;

  // A fragments for this wave's 16 rows, all K-steps, held in registers.
  v16h afr[KF];
  const h16* arow = A + (size_t)(mbase + lmod) * KPAD;
#pragma unroll
  for (int kf = 0; kf < KF; kf++) {
    v8h lo = *(const v8h*)(arow + kf * 32 + lhalf * 16);
    v8h hi = *(const v8h*)(arow + kf * 32 + lhalf * 16 + 8);
    v16h f;
#pragma unroll
    for (int i = 0; i < 8; i++) { f[i] = lo[i]; f[8 + i] = hi[i]; }
    afr[kf] = f;
  }

  int ntiles = coutP >> 4;
  for (int nt = 0; nt < ntiles; nt++) {
    int n = nt * 16 + lmod;   // this lane's output column (channel)
    v8f acc = {};
#pragma unroll
    for (int kf = 0; kf < KF; kf++) {
      v16h bf = *(const v16h*)(sW + n * KPAD + kf * 32 + lhalf * 16);
      acc = __builtin_amdgcn_wmma_f32_16x16x32_f16(
          false, afr[kf], false, bf, (short)0, acc, false, false);
    }
    int m0 = mbase + lhalf * 8;  // acc[r] is row m0+r, column n
    if (EPI == 0) {
      if (n < cout) {
        v8h o;
#pragma unroll
        for (int r = 0; r < 8; r++) o[r] = (h16)acc[r];
        *(v8h*)(outP + (size_t)n * MTOT + m0) = o;
      }
    } else if (EPI == 1) {
#pragma unroll
      for (int r = 0; r < 8; r++)
        outF[(size_t)(m0 + r) * CCH + n] =
            acc[r] + res[(size_t)(m0 + r) * CCH + n];
    } else {  // EPI == 2: final output, NCHW layout + residual
      int b = m0 / NPIX, pix = m0 % NPIX;
      float* ob = outF + ((size_t)b * CCH + n) * NPIX + pix;
#pragma unroll
      for (int r = 0; r < 8; r++)
        ob[r] = acc[r] + res[(size_t)(m0 + r) * CCH + n];
    }
  }
}

// ---------------------------------------------------------------------------
// Depthwise 3x3 conv over channel planes (pad=1, no bias).
// Planes are [c][b*NPIX + h*WW + w].  One block = one image row (256 threads).
// ---------------------------------------------------------------------------
__global__ __launch_bounds__(256) void k_dw3x3(const h16* __restrict__ in,
                                               const float* __restrict__ wt,
                                               h16* __restrict__ out) {
  int row = blockIdx.x;
  int c   = row / (BATCH * HH);
  int rem = row % (BATCH * HH);
  int b = rem / HH, hh = rem % HH;
  int wx = threadIdx.x;
  const float* wk = wt + c * 9;
  const h16* ip = in + (size_t)c * MTOT + (size_t)b * NPIX;
  if (hh + 2 < HH) __builtin_prefetch(ip + (hh + 2) * WW + wx, 0, 1);
  float acc = 0.f;
#pragma unroll
  for (int dy = -1; dy <= 1; dy++) {
    int y = hh + dy;
    if ((unsigned)y >= (unsigned)HH) continue;
#pragma unroll
    for (int dx = -1; dx <= 1; dx++) {
      int xq = wx + dx;
      if ((unsigned)xq >= (unsigned)WW) continue;
      acc += wk[(dy + 1) * 3 + (dx + 1)] * (float)ip[y * WW + xq];
    }
  }
  out[(size_t)c * MTOT + (size_t)b * NPIX + hh * WW + wx] = (h16)acc;
}

// ---------------------------------------------------------------------------
// Grouped 3x3 conv (groups=64, 2 in-ch per group): x1 branch.
// in: dw1 planes (128), out: 64 planes.
// ---------------------------------------------------------------------------
__global__ __launch_bounds__(256) void k_gconv_x1(const h16* __restrict__ in,
                                                  const float* __restrict__ wt,
                                                  h16* __restrict__ out) {
  int row = blockIdx.x;
  int c   = row / (BATCH * HH);
  int rem = row % (BATCH * HH);
  int b = rem / HH, hh = rem % HH;
  int wx = threadIdx.x;
  float acc = 0.f;
#pragma unroll
  for (int g = 0; g < 2; g++) {
    const h16* ip = in + (size_t)(2 * c + g) * MTOT + (size_t)b * NPIX;
    const float* wk = wt + (c * 2 + g) * 9;
#pragma unroll
    for (int dy = -1; dy <= 1; dy++) {
      int y = hh + dy;
      if ((unsigned)y >= (unsigned)HH) continue;
#pragma unroll
      for (int dx = -1; dx <= 1; dx++) {
        int xq = wx + dx;
        if ((unsigned)xq >= (unsigned)WW) continue;
        acc += wk[(dy + 1) * 3 + (dx + 1)] * (float)ip[y * WW + xq];
      }
    }
  }
  out[(size_t)c * MTOT + (size_t)b * NPIX + hh * WW + wx] = (h16)acc;
}

// ---------------------------------------------------------------------------
// L2-normalize q and k over the per-head channel dim (8), in place.
// qk planes: 0..63 = q, 64..127 = k (all heads).
// ---------------------------------------------------------------------------
__global__ __launch_bounds__(256) void k_qknorm(h16* __restrict__ qk) {
  size_t m = (size_t)blockIdx.x * 256 + threadIdx.x;
#pragma unroll
  for (int hd = 0; hd < NHEAD; hd++) {
    float qv[CHD], kv[CHD];
    float qs = 0.f, ks = 0.f;
#pragma unroll
    for (int c = 0; c < CHD; c++) {
      qv[c] = (float)qk[(size_t)(hd * CHD + c) * MTOT + m];
      kv[c] = (float)qk[(size_t)(64 + hd * CHD + c) * MTOT + m];
      qs += qv[c] * qv[c];
      ks += kv[c] * kv[c];
    }
    float qf = 1.0f / (sqrtf(qs) + 1e-6f);
    float kf = 1.0f / (sqrtf(ks) + 1e-6f);
#pragma unroll
    for (int c = 0; c < CHD; c++) {
      qk[(size_t)(hd * CHD + c) * MTOT + m] = (h16)(qv[c] * qf);
      qk[(size_t)(64 + hd * CHD + c) * MTOT + m] = (h16)(kv[c] * kf);
    }
  }
}

// ---------------------------------------------------------------------------
// Refine-attention per-head conv (window 3/5/7), 16 input planes (q,k of the
// head), 1 output plane per head, sigmoid applied.  grid=(BATCH*HH, 8 heads).
// ---------------------------------------------------------------------------
__global__ __launch_bounds__(256) void k_refine(
    const h16* __restrict__ qkd, const float* __restrict__ r3w,
    const float* __restrict__ r3b, const float* __restrict__ r5w,
    const float* __restrict__ r5b, const float* __restrict__ r7w,
    const float* __restrict__ r7b, h16* __restrict__ att) {
  int hd  = blockIdx.y;
  int row = blockIdx.x;
  int b = row / HH, hh = row % HH;
  int wx = threadIdx.x;
  int win; const float* wk; float bias;
  if (hd < 2)      { win = 3; wk = r3w + hd * 16 * 9;        bias = r3b[hd]; }
  else if (hd < 5) { win = 5; wk = r5w + (hd - 2) * 16 * 25; bias = r5b[hd - 2]; }
  else             { win = 7; wk = r7w + (hd - 5) * 16 * 49; bias = r7b[hd - 5]; }
  int rad = win >> 1;
  __shared__ float sw[16 * 49];
  for (int i = threadIdx.x; i < 16 * win * win; i += 256) sw[i] = wk[i];
  __syncthreads();
  float acc = bias;
  for (int j = 0; j < 16; j++) {
    int plane = (j < 8) ? (hd * CHD + j) : (64 + hd * CHD + j - 8);
    const h16* ip = qkd + (size_t)plane * MTOT + (size_t)b * NPIX;
    const float* wj = sw + j * win * win;
    for (int dy = -rad; dy <= rad; dy++) {
      int y = hh + dy;
      if ((unsigned)y >= (unsigned)HH) continue;
      for (int dx = -rad; dx <= rad; dx++) {
        int xq = wx + dx;
        if ((unsigned)xq >= (unsigned)WW) continue;
        acc += wj[(dy + rad) * win + (dx + rad)] * (float)ip[y * WW + xq];
      }
    }
  }
  float sg = 1.0f / (1.0f + expf(-acc));
  att[(size_t)hd * MTOT + (size_t)b * NPIX + hh * WW + wx] = (h16)sg;
}

// ---------------------------------------------------------------------------
// Per-(batch,head) reductions: attn[8][8] = k@v, ksum[8], vsum[8].
// tables layout per (b,hd): [attn 64 | ksum 8 | vsum 8] f32 (pre-zeroed).
// grid = (chunks=64, BATCH*NHEAD=32).  Wave32 shuffle reduce + atomics.
// ---------------------------------------------------------------------------
__global__ __launch_bounds__(256) void k_kv_reduce(const h16* __restrict__ qkvd,
                                                   float* __restrict__ tables) {
  int bh = blockIdx.y;
  int b = bh / NHEAD, hd = bh % NHEAD;
  float acc[80];
#pragma unroll
  for (int t = 0; t < 80; t++) acc[t] = 0.f;
  const h16* kp = qkvd + (size_t)(64 + hd * CHD) * MTOT + (size_t)b * NPIX;
  const h16* vp = qkvd + (size_t)(128 + hd * CHD) * MTOT + (size_t)b * NPIX;
  for (int pix = blockIdx.x * 256 + threadIdx.x; pix < NPIX;
       pix += gridDim.x * 256) {
    float kv[CHD], vv[CHD];
#pragma unroll
    for (int i = 0; i < CHD; i++) {
      kv[i] = (float)kp[(size_t)i * MTOT + pix];
      vv[i] = (float)vp[(size_t)i * MTOT + pix];
    }
#pragma unroll
    for (int i = 0; i < CHD; i++) {
#pragma unroll
      for (int j = 0; j < CHD; j++) acc[i * 8 + j] += kv[i] * vv[j];
      acc[64 + i] += kv[i];
      acc[72 + i] += vv[i];
    }
  }
  float* tb = tables + bh * 80;
#pragma unroll
  for (int t = 0; t < 80; t++) {
    float v = acc[t];
#pragma unroll
    for (int off = 16; off > 0; off >>= 1) v += __shfl_down(v, off, 32);
    if ((threadIdx.x & 31) == 0) atomicAdd(tb + t, v);
  }
}

// ---------------------------------------------------------------------------
// Attention epilogue per pixel: out = (vsum + q@attn)/(N + q@ksum + eps)
//   * temperature * sigmoid(refine) * xsca * gelu(x1)  -> pixel-major f16 A1.
// ---------------------------------------------------------------------------
__global__ __launch_bounds__(256) void k_attn_out(
    const h16* __restrict__ qkvd, const h16* __restrict__ x1p,
    const h16* __restrict__ att, const float* __restrict__ tables,
    const float* __restrict__ xsca, const float* __restrict__ temp,
    h16* __restrict__ A1) {
  size_t m = (size_t)blockIdx.x * 256 + threadIdx.x;
  int b = (int)(m / NPIX);
  __shared__ float stb[NHEAD * 80];
  __shared__ float ssca[CCH];
  __shared__ float stemp[NHEAD];
  for (int i = threadIdx.x; i < NHEAD * 80; i += 256)
    stb[i] = tables[b * NHEAD * 80 + i];
  if (threadIdx.x < CCH) ssca[threadIdx.x] = xsca[b * CCH + threadIdx.x];
  if (threadIdx.x < NHEAD) stemp[threadIdx.x] = temp[threadIdx.x];
  __syncthreads();
  h16* orow = A1 + m * CCH;
#pragma unroll
  for (int hd = 0; hd < NHEAD; hd++) {
    const float* tb = stb + hd * 80;
    float qv[CHD];
    float den = (float)NPIX + 1e-6f;
#pragma unroll
    for (int c = 0; c < CHD; c++) {
      qv[c] = (float)qkvd[(size_t)(hd * CHD + c) * MTOT + m];
      den += qv[c] * tb[64 + c];
    }
    float rw    = (float)att[(size_t)hd * MTOT + m];
    float scale = stemp[hd] * rw / den;
#pragma unroll
    for (int j = 0; j < CHD; j++) {
      float num = tb[72 + j];
#pragma unroll
      for (int c = 0; c < CHD; c++) num += qv[c] * tb[c * 8 + j];
      int ch    = hd * CHD + j;
      float x1v = (float)x1p[(size_t)ch * MTOT + m];
      float g   = 0.5f * x1v * (1.0f + erff(x1v * 0.70710678f));
      orow[ch]  = (h16)(num * scale * ssca[ch] * g);
    }
  }
}

// ---------------------------------------------------------------------------
// SCA: spatial mean of LN1 output per (b,c) -> sums (atomic), then tiny GEMV.
// ---------------------------------------------------------------------------
__global__ __launch_bounds__(256) void k_sca_reduce(const h16* __restrict__ A0,
                                                    float* __restrict__ sums) {
  int b = blockIdx.y;
  float acc[CCH];
#pragma unroll
  for (int c = 0; c < CCH; c++) acc[c] = 0.f;
  for (int pix = blockIdx.x * 256 + threadIdx.x; pix < NPIX;
       pix += gridDim.x * 256) {
    const h16* r = A0 + ((size_t)b * NPIX + pix) * CCH;
#pragma unroll
    for (int c = 0; c < CCH; c++) acc[c] += (float)r[c];
  }
#pragma unroll
  for (int c = 0; c < CCH; c++) {
    float v = acc[c];
#pragma unroll
    for (int off = 16; off > 0; off >>= 1) v += __shfl_down(v, off, 32);
    if ((threadIdx.x & 31) == 0) atomicAdd(sums + b * CCH + c, v);
  }
}

__global__ __launch_bounds__(256) void k_sca_gemv(const float* __restrict__ sums,
                                                  const float* __restrict__ w,
                                                  const float* __restrict__ bias,
                                                  float* __restrict__ xsca) {
  int b = threadIdx.x / CCH;   // 256 threads = 4 batches x 64 channels
  int c = threadIdx.x % CCH;
  float acc = bias[c];
  for (int ci = 0; ci < CCH; ci++)
    acc += w[c * CCH + ci] * (sums[b * CCH + ci] * (1.0f / NPIX));
  xsca[b * CCH + c] = acc;
}

// ---------------------------------------------------------------------------
// FFN activation: gelu(h1) * h2, planar -> pixel-major (K padded 170 -> 192).
// ---------------------------------------------------------------------------
__global__ __launch_bounds__(256) void k_ffn_act(const h16* __restrict__ hdp,
                                                 h16* __restrict__ A3) {
  size_t m = (size_t)blockIdx.x * 256 + threadIdx.x;
  h16* o = A3 + m * 192;
  for (int c = 0; c < HIDD; c++) {
    float a  = (float)hdp[(size_t)c * MTOT + m];
    float b2 = (float)hdp[(size_t)(c + HIDD) * MTOT + m];
    float g  = 0.5f * a * (1.0f + erff(a * 0.70710678f));
    o[c] = (h16)(g * b2);
  }
#pragma unroll
  for (int c = HIDD; c < 192; c++) o[c] = (h16)0.0f;
}

// ---------------------------------------------------------------------------
// Host launcher
// ---------------------------------------------------------------------------
extern "C" void kernel_launch(void* const* d_in, const int* in_sizes, int n_in,
                              void* d_out, int out_size, void* d_ws,
                              size_t ws_size, hipStream_t stream) {
  (void)in_sizes; (void)n_in; (void)out_size; (void)ws_size;
  const float* x        = (const float*)d_in[0];
  const float* ln1_w    = (const float*)d_in[1];
  const float* sca_w    = (const float*)d_in[2];
  const float* sca_b    = (const float*)d_in[3];
  const float* dw1_w    = (const float*)d_in[4];
  const float* dw2_w    = (const float*)d_in[5];
  const float* qkv_w    = (const float*)d_in[6];
  const float* qkv_dw_w = (const float*)d_in[7];
  const float* temp     = (const float*)d_in[8];
  const float* r3w      = (const float*)d_in[9];
  const float* r3b      = (const float*)d_in[10];
  const float* r5w      = (const float*)d_in[11];
  const float* r5b      = (const float*)d_in[12];
  const float* r7w      = (const float*)d_in[13];
  const float* r7b      = (const float*)d_in[14];
  const float* proj_w   = (const float*)d_in[15];
  const float* ln2_w    = (const float*)d_in[16];
  const float* pin_w    = (const float*)d_in[17];
  const float* ffn_dw_w = (const float*)d_in[18];
  const float* pout_w   = (const float*)d_in[19];
  float* out = (float*)d_out;

  // Workspace carve-out (lifetime-aliased regions).
  char* wsp = (char*)d_ws;
  size_t off = 0;
  auto alloc = [&](size_t bytes) -> void* {
    void* p = wsp + off;
    off += (bytes + 255) & ~(size_t)255;
    return p;
  };
  h16*   A0     = (h16*)alloc((size_t)MTOT * CCH * 2);   // LN1 out; reused as LN2 out
  h16*   A1     = (h16*)alloc((size_t)MTOT * CCH * 2);   // attention out (pixel-major)
  float* xi     = (float*)alloc((size_t)MTOT * CCH * 4); // residual after attention
  h16*   attp   = (h16*)alloc((size_t)NHEAD * MTOT * 2); // refine maps (sigmoid)
  h16*   R1     = (h16*)alloc((size_t)352 * MTOT * 2);   // qkv+dw1 planes / pin planes / A3
  h16*   R2     = (h16*)alloc((size_t)352 * MTOT * 2);   // qkvd(192)+x1(64) / ffn-dw planes
  h16*   Wc     = (h16*)alloc((size_t)320 * 64 * 2);     // [qkv_w ; dw1_w]
  h16*   Wproj  = (h16*)alloc((size_t)64 * 64 * 2);
  h16*   Wpin   = (h16*)alloc((size_t)352 * 64 * 2);
  h16*   Wpout  = (h16*)alloc((size_t)64 * 192 * 2);
  float* sums   = (float*)alloc(BATCH * CCH * 4);
  float* xsca   = (float*)alloc(BATCH * CCH * 4);
  float* tables = (float*)alloc(BATCH * NHEAD * 80 * 4);

  const int TPB = 256;
  auto g1 = [](int n) { return (n + 255) / 256; };

  // Weight conversion / packing (fp16, zero-padded as needed).
  k_cvt_pad<<<g1(192 * 64), TPB, 0, stream>>>(qkv_w, Wc, 192, 64, 192, 64);
  k_cvt_pad<<<g1(128 * 64), TPB, 0, stream>>>(dw1_w, Wc + 192 * 64, 128, 64, 128, 64);
  k_cvt_pad<<<g1(64 * 64), TPB, 0, stream>>>(proj_w, Wproj, 64, 64, 64, 64);
  k_cvt_pad<<<g1(352 * 64), TPB, 0, stream>>>(pin_w, Wpin, 340, 64, 352, 64);
  k_cvt_pad<<<g1(64 * 192), TPB, 0, stream>>>(pout_w, Wpout, 64, 170, 64, 192);

  // ---- Attention branch ----
  k_ln<<<MTOT / TPB, TPB, 0, stream>>>(x, ln1_w, A0);

  k_zero_f32<<<g1(BATCH * CCH), TPB, 0, stream>>>(sums, BATCH * CCH);
  k_zero_f32<<<g1(BATCH * NHEAD * 80), TPB, 0, stream>>>(tables, BATCH * NHEAD * 80);
  k_sca_reduce<<<dim3(32, BATCH), TPB, 0, stream>>>(A0, sums);
  k_sca_gemv<<<1, TPB, 0, stream>>>(sums, sca_w, sca_b, xsca);

  // Fused qkv(192) + dw1(128) GEMM -> channel planes in R1.
  k_gemm_wmma<64, 0><<<MTOT / 128, TPB, 0, stream>>>(A0, Wc, 320, 320, R1,
                                                     nullptr, nullptr);
  // Depthwise 3x3 on qkv planes -> R2[0..191].
  k_dw3x3<<<192 * BATCH * HH, TPB, 0, stream>>>(R1, qkv_dw_w, R2);
  // Grouped 3x3 on dw1 planes -> x1 planes R2[192..255].
  k_gconv_x1<<<CCH * BATCH * HH, TPB, 0, stream>>>(R1 + (size_t)192 * MTOT,
                                                   dw2_w, R2 + (size_t)192 * MTOT);
  // Normalize q,k (in place), refine maps, k@v reductions.
  k_qknorm<<<MTOT / TPB, TPB, 0, stream>>>(R2);
  k_refine<<<dim3(BATCH * HH, NHEAD), TPB, 0, stream>>>(R2, r3w, r3b, r5w, r5b,
                                                        r7w, r7b, attp);
  k_kv_reduce<<<dim3(64, BATCH * NHEAD), TPB, 0, stream>>>(R2, tables);
  // Attention epilogue -> pixel-major A1.
  k_attn_out<<<MTOT / TPB, TPB, 0, stream>>>(R2, R2 + (size_t)192 * MTOT, attp,
                                             tables, xsca, temp, A1);
  // proj GEMM + residual -> xi (f32 pixel-major).
  k_gemm_wmma<64, 1><<<MTOT / 128, TPB, 0, stream>>>(A1, Wproj, 64, 64, nullptr,
                                                     xi, x);

  // ---- FFN branch ----
  k_ln<<<MTOT / TPB, TPB, 0, stream>>>(xi, ln2_w, A0);
  k_gemm_wmma<64, 0><<<MTOT / 128, TPB, 0, stream>>>(A0, Wpin, 352, 340, R1,
                                                     nullptr, nullptr);
  k_dw3x3<<<340 * BATCH * HH, TPB, 0, stream>>>(R1, ffn_dw_w, R2);
  k_ffn_act<<<MTOT / TPB, TPB, 0, stream>>>(R2, R1);  // A3 aliases R1
  // pout GEMM + residual -> final NCHW f32 output.
  k_gemm_wmma<192, 2><<<MTOT / 128, TPB, 0, stream>>>(R1, Wpout, 64, 64,
                                                      nullptr, out, xi);
}
